// MultiHeadAttention_19069654794357
// MI455X (gfx1250) — compile-verified, matrix operand order checked by reading, not run on hardware
//
#include <hip/hip_runtime.h>
#include <hip/hip_bf16.h>
#include <math.h>

// ---------------------------------------------------------------------------
// Causal multi-head attention forward for MI455X (gfx1250, wave32, WMMA).
// All GEMMs use v_wmma_f32_16x16x32_bf16 (fp32 accum). Ping-pong software
// pipelining (no register rotation copies), one lane-swizzled base pointer
// per operand so fragment addresses fold into load immediate offsets,
// WGP-scope global_prefetch_b8 on the streamed operands, and
// __launch_bounds__ so the register allocator never spills the hot loops.
// ---------------------------------------------------------------------------

typedef __bf16 bf16;
typedef __bf16 v16bf __attribute__((ext_vector_type(16)));
typedef __bf16 v8bf  __attribute__((ext_vector_type(8)));
typedef float  v8f   __attribute__((ext_vector_type(8)));

#define B_  4
#define S_  2048
#define E_  1024
#define H_  16
#define DK_ 64
#define DV_ 128
#define HD_ (H_ * DV_)   // 2048

// ---- WMMA helpers ---------------------------------------------------------

__device__ __forceinline__ v8f wmma_bf16(v16bf a, v16bf b, v8f c) {
  return __builtin_amdgcn_wmma_f32_16x16x32_bf16(
      /*neg_a=*/false, a, /*neg_b=*/false, b,
      /*c_mod=*/(short)0, c, /*reuse_a=*/false, /*reuse_b=*/false);
}

// Per-lane swizzled base pointer for 16x32 bf16 fragments of a row-major
// tile with leading dimension ld: lane m=lane&15 reads row m, half=lane>>4
// selects the two 16B chunks at +h*8 and +h*8+16 elements.
__device__ __forceinline__ const bf16* lane_ptr(const bf16* tile, int ld) {
  const int lane = threadIdx.x & 31;
  return tile + (size_t)(lane & 15) * ld + (lane >> 4) * 8;
}

// Load a fragment given the pre-swizzled lane pointer (+ element offset).
__device__ __forceinline__ v16bf load_frag_at(const bf16* lanep) {
  v8bf lo = *(const v8bf*)(lanep);
  v8bf hi = *(const v8bf*)(lanep + 16);
  v16bf r;
#pragma unroll
  for (int i = 0; i < 8; ++i) { r[i] = lo[i]; r[i + 8] = hi[i]; }
  return r;
}

// WGP-scope prefetch (locality 3 -> SCOPE_WGP: fill all cache levels).
__device__ __forceinline__ void wgp_prefetch(const bf16* p) {
  __builtin_prefetch((const void*)p, 0, 3);
}

// ---- Conversion kernels ---------------------------------------------------

__global__ void mha_cvt_bf16(const float* __restrict__ in, bf16* __restrict__ out, int n) {
  int i = blockIdx.x * blockDim.x + threadIdx.x;
  if (i < n) out[i] = (bf16)in[i];
}

// in: [Hn][R][C] f32 -> out: [Hn][C][R] bf16
__global__ void mha_cvt_transpose(const float* __restrict__ in, bf16* __restrict__ out,
                                  int Hn, int R, int C) {
  int i = blockIdx.x * blockDim.x + threadIdx.x;
  int total = Hn * R * C;
  if (i >= total) return;
  int c = i % C;
  int r = (i / C) % R;
  int hh = i / (C * R);
  out[((size_t)hh * C + c) * R + r] = (bf16)in[i];
}

// ---- Per-head projection GEMM: out[b,h,s,d] = x[b,s,:] @ W[h,:,d] + bias --
// Wt: [H][D][E] (transposed weights, bf16). One wave -> 32x64 output tile,
// ping-pong double-buffered over the K (=E) dimension.
// transposed==1 stores out as [B,H,D,S] (V becomes the pre-transposed B
// operand of the P@V GEMM).

__global__ __launch_bounds__(128, 1)
void mha_proj(const bf16* __restrict__ xbf,
              const bf16* __restrict__ Wt,
              const float* __restrict__ bias,
              bf16* __restrict__ out,
              int D, int transposed) {
  const int wave = blockIdx.x * (blockDim.x >> 5) + (threadIdx.x >> 5);
  const int lane = threadIdx.x & 31;
  const int ncol = D >> 6;
  const int nqt = S_ >> 5;                 // 32-row tiles
  const int col64 = wave % ncol;
  const int qt = (wave / ncol) % nqt;
  const int h  = (wave / (ncol * nqt)) % H_;
  const int b  =  wave / (ncol * nqt * H_);
  const int qbase = qt * 32;

  const bf16* A  = xbf + ((size_t)b * S_ + qbase) * E_;
  const bf16* Bw = Wt + (size_t)h * D * E_ + (size_t)(col64 * 64) * E_;
  const bf16* pA = lane_ptr(A, E_);    // rows 0-15; rows 16-31 at +16*E_
  const bf16* pB = lane_ptr(Bw, E_);   // col frag f at +f*16*E_

  v8f acc0[4] = {}, acc1[4] = {};

  // ping-pong fragment sets
  v16bf aA0, aA1, bA[4], aB0, aB1, bB[4];
  aA0 = load_frag_at(pA);
  aA1 = load_frag_at(pA + (size_t)16 * E_);
#pragma unroll
  for (int f = 0; f < 4; ++f) bA[f] = load_frag_at(pB + (size_t)(f * 16) * E_);

  const int nsteps = E_ / 32;              // 32 (even)
  for (int it = 0; it < nsteps; it += 2) {
    {
      const int en = (it + 1) * 32;        // always < E_
      aB0 = load_frag_at(pA + en);
      aB1 = load_frag_at(pA + (size_t)16 * E_ + en);
#pragma unroll
      for (int f = 0; f < 4; ++f) bB[f] = load_frag_at(pB + (size_t)(f * 16) * E_ + en);
      const int ep = en + 96;
      if (ep < E_) {
        wgp_prefetch(pA + ep);
        wgp_prefetch(pA + (size_t)16 * E_ + ep);
      }
    }
#pragma unroll
    for (int f = 0; f < 4; ++f) {
      acc0[f] = wmma_bf16(aA0, bA[f], acc0[f]);
      acc1[f] = wmma_bf16(aA1, bA[f], acc1[f]);
    }
    if (it + 2 < nsteps) {
      const int en = (it + 2) * 32;
      aA0 = load_frag_at(pA + en);
      aA1 = load_frag_at(pA + (size_t)16 * E_ + en);
#pragma unroll
      for (int f = 0; f < 4; ++f) bA[f] = load_frag_at(pB + (size_t)(f * 16) * E_ + en);
    }
#pragma unroll
    for (int f = 0; f < 4; ++f) {
      acc0[f] = wmma_bf16(aB0, bB[f], acc0[f]);
      acc1[f] = wmma_bf16(aB1, bB[f], acc1[f]);
    }
  }

  const int hh = lane >> 4, n = lane & 15;
#pragma unroll
  for (int g = 0; g < 2; ++g) {
    v8f* acc = g ? acc1 : acc0;
#pragma unroll
    for (int f = 0; f < 4; ++f) {
#pragma unroll
      for (int r = 0; r < 8; ++r) {
        const int M = g * 16 + r + 8 * hh;
        const int col = col64 * 64 + f * 16 + n;
        float v = acc[f][r] + bias[h * D + col];
        size_t offo;
        if (!transposed)
          offo = (((size_t)(b * H_ + h)) * S_ + qbase + M) * D + col;
        else
          offo = (((size_t)(b * H_ + h)) * D + col) * S_ + qbase + M;
        out[offo] = (bf16)v;
      }
    }
  }
}

// ---- Flash attention: one wave owns one (b,h,16-query tile) ---------------
// Q,K: [B,H,S,DK] bf16 ; Vt: [B,H,DV,S] bf16 ; Cc: [B,S,H*DV] bf16 (concat).

__global__ __launch_bounds__(128, 1)
void mha_attn(const bf16* __restrict__ Q,
              const bf16* __restrict__ K,
              const bf16* __restrict__ Vt,
              bf16* __restrict__ Cc) {
  __shared__ __align__(16) bf16 plds_all[4][16 * 48];
  const int wslot = threadIdx.x >> 5;
  bf16* plds = plds_all[wslot];
  const int wave = blockIdx.x * (blockDim.x >> 5) + wslot;
  const int lane = threadIdx.x & 31;
  const int nqt = S_ >> 4;
  const int qt = wave % nqt;
  const int h  = (wave / nqt) % H_;
  const int b  =  wave / (nqt * H_);
  const int qbase = qt * 16;
  const int hh = lane >> 4, n = lane & 15;

  const bf16* Qp = Q + (((size_t)(b * H_ + h)) * S_ + qbase) * DK_;
  const bf16* Kp = K + ((size_t)(b * H_ + h)) * S_ * DK_;
  const bf16* Vp = Vt + ((size_t)(b * H_ + h)) * DV_ * S_;

  const bf16* pQ = lane_ptr(Qp, DK_);
  const bf16* pK = lane_ptr(Kp, DK_);
  const bf16* pV = lane_ptr(Vp, S_);

  const v16bf qf0 = load_frag_at(pQ);
  const v16bf qf1 = load_frag_at(pQ + 32);

  v8f acc[8] = {};
  float mrow[8], lrow[8];
#pragma unroll
  for (int r = 0; r < 8; ++r) { mrow[r] = -__builtin_inff(); lrow[r] = 0.f; }

  const float sc = 0.125f;  // 1/sqrt(DK)
  const int kend = qbase + 16;

  // prologue: fragments for the first key block
  v16bf k0a = load_frag_at(pK);
  v16bf k0b = load_frag_at(pK + 32);
  v16bf k1a = load_frag_at(pK + (size_t)16 * DK_);
  v16bf k1b = load_frag_at(pK + (size_t)16 * DK_ + 32);

  for (int k0 = 0; k0 < kend; k0 += 32) {
    // ---- scores for 32 keys as two 16-key fragments (K dim = DK = 64) ----
    v8f s0 = {}, s1 = {};
    s0 = wmma_bf16(qf0, k0a, s0);
    s0 = wmma_bf16(qf1, k0b, s0);
    s1 = wmma_bf16(qf0, k1a, s1);
    s1 = wmma_bf16(qf1, k1b, s1);

    // issue V-fragment loads for this block now; consumed after softmax
    v16bf vf[8];
#pragma unroll
    for (int f = 0; f < 8; ++f) vf[f] = load_frag_at(pV + (size_t)(f * 16) * S_ + k0);

    // issue next key block fragment loads (overlap with softmax VALU)
    if (k0 + 32 < kend) {
      const bf16* kp = pK + (size_t)(k0 + 32) * DK_;
      k0a = load_frag_at(kp);
      k0b = load_frag_at(kp + 32);
      k1a = load_frag_at(kp + (size_t)16 * DK_);
      k1b = load_frag_at(kp + (size_t)16 * DK_ + 32);
    }
    // prefetch the key block two iterations ahead into the WGP cache
    if (k0 + 64 < kend) {
      wgp_prefetch(pK + (size_t)(k0 + 64) * DK_);
      wgp_prefetch(pK + (size_t)(k0 + 80) * DK_);
    }

    // ---- scale + causal mask ----
#pragma unroll
    for (int r = 0; r < 8; ++r) {
      const int qi = qbase + r + 8 * hh;
      float v0 = s0[r] * sc;
      float v1 = s1[r] * sc;
      s0[r] = ((k0 + n) > qi) ? -__builtin_inff() : v0;
      s1[r] = ((k0 + 16 + n) > qi) ? -__builtin_inff() : v1;
    }

    // ---- online softmax: row max, exp, row sum, rescale ----
    float rs[8];
#pragma unroll
    for (int r = 0; r < 8; ++r) {
      float v = fmaxf(s0[r], s1[r]);
      v = fmaxf(v, __shfl_xor(v, 8, 32));
      v = fmaxf(v, __shfl_xor(v, 4, 32));
      v = fmaxf(v, __shfl_xor(v, 2, 32));
      v = fmaxf(v, __shfl_xor(v, 1, 32));
      const float mnew = fmaxf(mrow[r], v);
      const float p0 = __expf(s0[r] - mnew);
      const float p1 = __expf(s1[r] - mnew);
      s0[r] = p0; s1[r] = p1;
      float rsum = p0 + p1;
      rsum += __shfl_xor(rsum, 8, 32);
      rsum += __shfl_xor(rsum, 4, 32);
      rsum += __shfl_xor(rsum, 2, 32);
      rsum += __shfl_xor(rsum, 1, 32);
      const float scale = __expf(mrow[r] - mnew);
      lrow[r] = lrow[r] * scale + rsum;
      mrow[r] = mnew;
      rs[r] = scale;
    }
#pragma unroll
    for (int f = 0; f < 8; ++f)
#pragma unroll
      for (int r = 0; r < 8; ++r)
        acc[f][r] *= rs[r];

    // ---- re-layout P (C-frag) -> A-frag via wave-private LDS ----
#pragma unroll
    for (int r = 0; r < 8; ++r) {
      plds[(r + 8 * hh) * 48 + n]      = (bf16)s0[r];
      plds[(r + 8 * hh) * 48 + 16 + n] = (bf16)s1[r];
    }
    const v16bf pf = load_frag_at(lane_ptr(plds, 48));

    // ---- acc += P @ V  (8 fragments of 16 DV columns) ----
#pragma unroll
    for (int f = 0; f < 8; ++f)
      acc[f] = wmma_bf16(pf, vf[f], acc[f]);
  }

  // ---- normalize and store into the concat buffer [B,S,H*DV] ----
  float inv[8];
#pragma unroll
  for (int r = 0; r < 8; ++r) inv[r] = 1.0f / lrow[r];
  bf16* outp = Cc + ((size_t)b * S_ + qbase) * HD_ + h * DV_;
#pragma unroll
  for (int f = 0; f < 8; ++f)
#pragma unroll
    for (int r = 0; r < 8; ++r) {
      const int M = r + 8 * hh;
      outp[(size_t)M * HD_ + f * 16 + n] = (bf16)(acc[f][r] * inv[r]);
    }
}

// ---- Output projection: out = Cc @ Wo + bo (f32 result) -------------------
// Cc: [B*S, 2048] bf16 ; Wot: [E][2048] bf16 ; out: [B*S, E] f32.
// One wave -> 32x64 tile, ping-pong double-buffered over K.

__global__ __launch_bounds__(128, 1)
void mha_outproj(const bf16* __restrict__ Cc,
                 const bf16* __restrict__ Wot,
                 const float* __restrict__ bo,
                 float* __restrict__ out) {
  const int wave = blockIdx.x * (blockDim.x >> 5) + (threadIdx.x >> 5);
  const int lane = threadIdx.x & 31;
  const int ncol = E_ >> 6;  // 16
  const int col64 = wave % ncol;
  const int rt = wave / ncol;               // 0 .. B*S/32-1
  const bf16* A  = Cc + (size_t)rt * 32 * HD_;
  const bf16* Bw = Wot + (size_t)(col64 * 64) * HD_;
  const bf16* pA = lane_ptr(A, HD_);
  const bf16* pB = lane_ptr(Bw, HD_);

  v8f acc0[4] = {}, acc1[4] = {};
  v16bf aA0, aA1, bA[4], aB0, aB1, bB[4];
  aA0 = load_frag_at(pA);
  aA1 = load_frag_at(pA + (size_t)16 * HD_);
#pragma unroll
  for (int f = 0; f < 4; ++f) bA[f] = load_frag_at(pB + (size_t)(f * 16) * HD_);

  const int nsteps = HD_ / 32;  // 64 (even)
  for (int it = 0; it < nsteps; it += 2) {
    {
      const int kn = (it + 1) * 32;
      aB0 = load_frag_at(pA + kn);
      aB1 = load_frag_at(pA + (size_t)16 * HD_ + kn);
#pragma unroll
      for (int f = 0; f < 4; ++f) bB[f] = load_frag_at(pB + (size_t)(f * 16) * HD_ + kn);
      const int kp = kn + 96;
      if (kp < HD_) {
        wgp_prefetch(pA + kp);
        wgp_prefetch(pA + (size_t)16 * HD_ + kp);
      }
    }
#pragma unroll
    for (int f = 0; f < 4; ++f) {
      acc0[f] = wmma_bf16(aA0, bA[f], acc0[f]);
      acc1[f] = wmma_bf16(aA1, bA[f], acc1[f]);
    }
    if (it + 2 < nsteps) {
      const int kn = (it + 2) * 32;
      aA0 = load_frag_at(pA + kn);
      aA1 = load_frag_at(pA + (size_t)16 * HD_ + kn);
#pragma unroll
      for (int f = 0; f < 4; ++f) bA[f] = load_frag_at(pB + (size_t)(f * 16) * HD_ + kn);
    }
#pragma unroll
    for (int f = 0; f < 4; ++f) {
      acc0[f] = wmma_bf16(aB0, bB[f], acc0[f]);
      acc1[f] = wmma_bf16(aB1, bB[f], acc1[f]);
    }
  }

  const int hh = lane >> 4, n = lane & 15;
#pragma unroll
  for (int g = 0; g < 2; ++g) {
    v8f* acc = g ? acc1 : acc0;
#pragma unroll
    for (int f = 0; f < 4; ++f)
#pragma unroll
      for (int r = 0; r < 8; ++r) {
        const int M = g * 16 + r + 8 * hh;
        const int col = col64 * 64 + f * 16 + n;
        out[((size_t)rt * 32 + M) * E_ + col] = acc[f][r] + bo[col];
      }
  }
}

// ---- Host-side launcher ---------------------------------------------------

extern "C" void kernel_launch(void* const* d_in, const int* in_sizes, int n_in,
                              void* d_out, int out_size, void* d_ws, size_t ws_size,
                              hipStream_t stream) {
  (void)in_sizes; (void)n_in; (void)out_size; (void)ws_size;

  const float* x  = (const float*)d_in[0];
  const float* Wq = (const float*)d_in[1];
  const float* bq = (const float*)d_in[2];
  const float* Wk = (const float*)d_in[3];
  const float* bk = (const float*)d_in[4];
  const float* Wv = (const float*)d_in[5];
  const float* bv = (const float*)d_in[6];
  const float* Wo = (const float*)d_in[7];
  const float* bo = (const float*)d_in[8];
  float* out = (float*)d_out;

  // Workspace carve-up (256B aligned).
  char* ws = (char*)d_ws;
  size_t off = 0;
  auto carve = [&](size_t bytes) -> char* {
    char* p = ws + off;
    off += (bytes + 255) & ~(size_t)255;
    return p;
  };
  bf16* xbf = (bf16*)carve((size_t)B_ * S_ * E_ * 2);
  bf16* Wqt = (bf16*)carve((size_t)H_ * DK_ * E_ * 2);
  bf16* Wkt = (bf16*)carve((size_t)H_ * DK_ * E_ * 2);
  bf16* Wvt = (bf16*)carve((size_t)H_ * DV_ * E_ * 2);
  bf16* Wot = (bf16*)carve((size_t)E_ * HD_ * 2);
  bf16* Qb  = (bf16*)carve((size_t)B_ * H_ * S_ * DK_ * 2);
  bf16* Kb  = (bf16*)carve((size_t)B_ * H_ * S_ * DK_ * 2);
  bf16* Vt  = (bf16*)carve((size_t)B_ * H_ * DV_ * S_ * 2);
  bf16* Cc  = (bf16*)carve((size_t)B_ * S_ * HD_ * 2);

  // 1) conversions / transposes
  {
    int n = B_ * S_ * E_;
    mha_cvt_bf16<<<(n + 255) / 256, 256, 0, stream>>>(x, xbf, n);
  }
  {
    int n = H_ * E_ * DK_;
    mha_cvt_transpose<<<(n + 255) / 256, 256, 0, stream>>>(Wq, Wqt, H_, E_, DK_);
    mha_cvt_transpose<<<(n + 255) / 256, 256, 0, stream>>>(Wk, Wkt, H_, E_, DK_);
  }
  {
    int n = H_ * E_ * DV_;
    mha_cvt_transpose<<<(n + 255) / 256, 256, 0, stream>>>(Wv, Wvt, H_, E_, DV_);
  }
  {
    int n = HD_ * E_;
    mha_cvt_transpose<<<(n + 255) / 256, 256, 0, stream>>>(Wo, Wot, 1, HD_, E_);
  }

  // 2) Q/K/V projections (wave = 32x64 tile; block = 4 waves)
  {
    int wavesQK = B_ * H_ * (S_ / 32) * (DK_ / 64);  // 4096
    mha_proj<<<wavesQK / 4, 128, 0, stream>>>(xbf, Wqt, bq, Qb, DK_, 0);
    mha_proj<<<wavesQK / 4, 128, 0, stream>>>(xbf, Wkt, bk, Kb, DK_, 0);
    int wavesV = B_ * H_ * (S_ / 32) * (DV_ / 64);   // 8192
    mha_proj<<<wavesV / 4, 128, 0, stream>>>(xbf, Wvt, bv, Vt, DV_, 1);
  }

  // 3) flash attention (wave = one 16-query tile)
  {
    int waves = B_ * H_ * (S_ / 16);  // 8192
    mha_attn<<<waves / 4, 128, 0, stream>>>(Qb, Kb, Vt, Cc);
  }

  // 4) output projection -> f32 (wave = 32x64 tile)
  {
    int waves = (B_ * S_ / 32) * (E_ / 64);  // 4096
    mha_outproj<<<waves / 4, 128, 0, stream>>>(Cc, Wot, bo, out);
  }
}
